// TransformerModel_66597762892596
// MI455X (gfx1250) — matrix operand order repaired and checked by reading, not compiled, not run on hardware
//
#include <hip/hip_runtime.h>
#include <cstdint>
#include <cstddef>

// ---------------- model constants ----------------
#define BB   2
#define TT   2048
#define DD   1024
#define HH   16
#define LL   4
#define DFF  4096
#define VV   32000
#define DKK  64
#define MTOK (BB * TT)   // 4096 token rows

typedef __attribute__((ext_vector_type(16))) __bf16 v16bf;
typedef __attribute__((ext_vector_type(8)))  float  v8f;
typedef __attribute__((ext_vector_type(8)))  __bf16 bf16x8;

// ---------------- gfx1250 async copy: global -> LDS (ASYNCcnt) ----------------
// GVS mode: mem_addr = SGPR64 + VGPR32_offset; LDS dest byte-offset in VGPR.
__device__ __forceinline__ void async_ld_b128(unsigned lds_addr, unsigned long long gbase, int goff) {
#if defined(__HIP_DEVICE_COMPILE__)
    asm volatile("global_load_async_to_lds_b128 %0, %1, %2"
                 :: "v"(lds_addr), "v"(goff), "s"(gbase)
                 : "memory");
#else
    (void)lds_addr; (void)gbase; (void)goff;
#endif
}
__device__ __forceinline__ void wait_async0() {
#if defined(__HIP_DEVICE_COMPILE__)
    asm volatile("s_wait_asynccnt 0" ::: "memory");
#endif
}

// ---------------- f32 -> bf16 conversion ----------------
__global__ __launch_bounds__(256)
void cvt_f32_bf16_kernel(const float* __restrict__ s, __bf16* __restrict__ d, long n) {
    long i = (long)blockIdx.x * blockDim.x + threadIdx.x;
    long stride = (long)gridDim.x * blockDim.x;
    for (; i < n; i += stride) d[i] = (__bf16)s[i];
}

// ---------------- embedding: x = wte[ids] + wpe ----------------
__global__ __launch_bounds__(256)
void embed_kernel(const int* __restrict__ ids, const float* __restrict__ wte,
                  const float* __restrict__ wpe, float* __restrict__ x) {
    long i = (long)blockIdx.x * 256 + threadIdx.x;      // over MTOK*DD
    int d = (int)(i & (DD - 1));
    long tok = i >> 10;                                  // DD = 1024
    int t = (int)(tok & (TT - 1));
    int id = ids[tok];
    x[i] = wte[(size_t)id * DD + d] + wpe[(size_t)t * DD + d];
}

// ---------------- LayerNorm (row of 1024) -> bf16 ----------------
__global__ __launch_bounds__(256)
void layernorm_bf16_kernel(const float* __restrict__ x, const float* __restrict__ g,
                           const float* __restrict__ bta, __bf16* __restrict__ out) {
    __shared__ float red[256];
    const int tid = threadIdx.x;
    const int row = blockIdx.x;
    const float* xr = x + (size_t)row * DD;
    float v0[4];
    float s = 0.f;
#pragma unroll
    for (int j = 0; j < 4; ++j) { v0[j] = xr[tid + j * 256]; s += v0[j]; }
    red[tid] = s; __syncthreads();
    for (int off = 128; off > 0; off >>= 1) {
        if (tid < off) red[tid] += red[tid + off];
        __syncthreads();
    }
    const float mean = red[0] * (1.f / DD);
    __syncthreads();
    float s2 = 0.f;
#pragma unroll
    for (int j = 0; j < 4; ++j) { float dl = v0[j] - mean; s2 += dl * dl; }
    red[tid] = s2; __syncthreads();
    for (int off = 128; off > 0; off >>= 1) {
        if (tid < off) red[tid] += red[tid + off];
        __syncthreads();
    }
    const float rstd = rsqrtf(red[0] * (1.f / DD) + 1e-5f);
#pragma unroll
    for (int j = 0; j < 4; ++j) {
        int c = tid + j * 256;
        out[(size_t)row * DD + c] = (__bf16)((v0[j] - mean) * rstd * g[c] + bta[c]);
    }
}

// ---------------- WMMA GEMM: C[M,N] = A[M,K](bf16) * W[N,K](bf16)^T ----------------
// Block tile 128x128, K-tile 64, double-buffered LDS pipeline:
//   iteration i: issue async tile i+1 into buf^1, WMMA on buf (overlaps DMA),
//   then s_wait_asynccnt 0 + barrier.
// 8 waves (2x4): each wave 64x32 = 4x2 WMMA tiles.
// EPI: 0 = store bf16; 1 = outF += acc; 2 = bf16(gelu(acc+bias)); 3 = outF += acc+bias; 4 = store f32
template <int EPI>
__global__ __launch_bounds__(256)
void gemm_bf16_wmma(const __bf16* __restrict__ A, const __bf16* __restrict__ W,
                    const float* __restrict__ bias, float* __restrict__ outF,
                    __bf16* __restrict__ outB, int M, int N, int K) {
    __shared__ __bf16 sA[2][128][72];
    __shared__ __bf16 sW[2][128][72];
    const int tid = threadIdx.x;
    const int lane = tid & 31;
    const int wv = tid >> 5;         // wave 0..7
    const int wm = wv >> 2;          // 0..1 -> 64-row strip
    const int wn = wv & 3;           // 0..3 -> 32-col strip
    const int hlf = lane >> 4;       // lane half (ISA A/B layout split)
    const int l15 = lane & 15;
    const int m0 = blockIdx.y * 128;
    const int n0 = blockIdx.x * 128;
    const unsigned long long Abase = (unsigned long long)(uintptr_t)A;
    const unsigned long long Wbase = (unsigned long long)(uintptr_t)W;
    // each thread owns 4 vec8 slots of each tile
    const int ldrow = tid >> 3;            // 0..31 (+32 per j)
    const int ldcol = (tid & 7) * 8;

    const v8f vzero = {};
    v8f acc[4][2];
#pragma unroll
    for (int mt = 0; mt < 4; ++mt)
#pragma unroll
        for (int nt = 0; nt < 2; ++nt) acc[mt][nt] = vzero;

    // prologue: stage tile 0 into buffer 0
#pragma unroll
    for (int j = 0; j < 4; ++j) {
        int row = ldrow + j * 32;
        async_ld_b128((unsigned)(uintptr_t)&sA[0][row][ldcol], Abase,
                      (int)(((size_t)(m0 + row) * K + ldcol) * 2));
        async_ld_b128((unsigned)(uintptr_t)&sW[0][row][ldcol], Wbase,
                      (int)(((size_t)(n0 + row) * K + ldcol) * 2));
    }
    wait_async0();
    __syncthreads();

    int buf = 0;
    for (int kt = 0; kt < K; kt += 64, buf ^= 1) {
        // issue next tile into the other buffer (overlaps with WMMA below)
        if (kt + 64 < K) {
#pragma unroll
            for (int j = 0; j < 4; ++j) {
                int row = ldrow + j * 32;
                async_ld_b128((unsigned)(uintptr_t)&sA[buf ^ 1][row][ldcol], Abase,
                              (int)(((size_t)(m0 + row) * K + kt + 64 + ldcol) * 2));
                async_ld_b128((unsigned)(uintptr_t)&sW[buf ^ 1][row][ldcol], Wbase,
                              (int)(((size_t)(n0 + row) * K + kt + 64 + ldcol) * 2));
            }
            // warm L2 for the tile after next (gfx1250 global_prefetch_b8)
            if (kt + 128 < K) {
                __builtin_prefetch(&A[(size_t)(m0 + ldrow) * K + kt + 128 + ldcol], 0, 0);
                __builtin_prefetch(&W[(size_t)(n0 + ldrow) * K + kt + 128 + ldcol], 0, 0);
            }
        }
#pragma unroll
        for (int ks = 0; ks < 2; ++ks) {
            v16bf af[4], bw[2];
#pragma unroll
            for (int mt = 0; mt < 4; ++mt) {
                int row = wm * 64 + mt * 16 + l15;
                int klo = hlf * 8;
#pragma unroll
                for (int v = 0; v < 8; ++v) {
                    int kk = ks * 32 + klo * 2 + 2 * v;
                    af[mt][2 * v]     = sA[buf][row][kk];
                    af[mt][2 * v + 1] = sA[buf][row][kk + 1];
                }
            }
#pragma unroll
            for (int nt = 0; nt < 2; ++nt) {
                int cn = wn * 32 + nt * 16 + l15;
                int kb0 = hlf * 16 + ks * 32;
#pragma unroll
                for (int v = 0; v < 8; ++v) {
                    bw[nt][2 * v]     = sW[buf][cn][kb0 + 2 * v];
                    bw[nt][2 * v + 1] = sW[buf][cn][kb0 + 2 * v + 1];
                }
            }
#pragma unroll
            for (int mt = 0; mt < 4; ++mt)
#pragma unroll
                for (int nt = 0; nt < 2; ++nt)
                    acc[mt][nt] = __builtin_amdgcn_wmma_f32_16x16x32_bf16(
                        false, af[mt], false, bw[nt], (short)0, acc[mt][nt], false, false);
        }
        // next tile fully in LDS + all waves done reading current buffer
        wait_async0();
        __syncthreads();
    }

    // epilogue (C layout: VGPR r of lane -> M = r + 8*half, N = l15)
#pragma unroll
    for (int mt = 0; mt < 4; ++mt)
#pragma unroll
        for (int nt = 0; nt < 2; ++nt)
#pragma unroll
            for (int r = 0; r < 8; ++r) {
                int m = m0 + wm * 64 + mt * 16 + r + hlf * 8;
                int n = n0 + wn * 32 + nt * 16 + l15;
                float v = acc[mt][nt][r];
                size_t idx = (size_t)m * N + n;
                if (EPI == 0) {
                    outB[idx] = (__bf16)v;
                } else if (EPI == 1) {
                    outF[idx] += v;
                } else if (EPI == 2) {
                    v += bias[n];
                    outB[idx] = (__bf16)(0.5f * v * (1.f + erff(v * 0.70710678f)));
                } else if (EPI == 3) {
                    outF[idx] += v + bias[n];
                } else {
                    outF[idx] = v;
                }
            }
}

// ---------------- flash attention (causal, DK=64) ----------------
// Block: 128 threads = 4 waves. Each block: one (b,h,q-tile of 64).
// Wave w owns q rows [w*16, w*16+16). WMMA for S = Q K^T and O += P V.
// Q/K staged via async-to-LDS; V transposed into sVt so P*V B-fragments
// are contiguous ds_load_b128 (no strided u16 loads).
__global__ __launch_bounds__(128)
void flash_attn_kernel(const __bf16* __restrict__ qkv, __bf16* __restrict__ o) {
    __shared__ __bf16 sQ[64][72];
    __shared__ __bf16 sK[64][72];
    __shared__ __bf16 sVt[64][72];   // sVt[d][kv]
    __shared__ __bf16 sP[64][72];
    const int tid = threadIdx.x;
    const int lane = tid & 31;
    const int wv = tid >> 5;     // 0..3
    const int hlf = lane >> 4;
    const int l15 = lane & 15;
    const int q0 = blockIdx.x * 64;
    const int b = blockIdx.y >> 4;   // / HH
    const int h = blockIdx.y & 15;
    const size_t baseQ = (size_t)b * TT * (3 * DD) + (size_t)h * DKK;
    const unsigned long long Qbase = (unsigned long long)(uintptr_t)qkv;
    const int ldrow = tid >> 3;          // 0..15 (+16 per j)
    const int ldcol = (tid & 7) * 8;

    // Q tile (async)
#pragma unroll
    for (int j = 0; j < 4; ++j) {
        int row = ldrow + j * 16;
        async_ld_b128((unsigned)(uintptr_t)&sQ[row][ldcol], Qbase,
                      (int)((baseQ + (size_t)(q0 + row) * (3 * DD) + ldcol) * 2));
    }
    float mr[8], lr[8];
    v8f oacc[4];
    const v8f vzero = {};
#pragma unroll
    for (int r = 0; r < 8; ++r) { mr[r] = -1e30f; lr[r] = 0.f; }
#pragma unroll
    for (int nt = 0; nt < 4; ++nt) oacc[nt] = vzero;

    const float scale = 0.125f;  // 1/sqrt(64)
    for (int k0 = 0; k0 <= q0; k0 += 64) {
#pragma unroll
        for (int j = 0; j < 4; ++j) {
            int row = ldrow + j * 16;
            size_t gb = baseQ + (size_t)(k0 + row) * (3 * DD) + ldcol;
            // K tile: async straight copy
            async_ld_b128((unsigned)(uintptr_t)&sK[row][ldcol], Qbase, (int)((gb + DD) * 2));
            // V tile: load to regs, transpose into sVt[d][kv]
            bf16x8 vv = *(const bf16x8*)&qkv[gb + 2 * DD];
#pragma unroll
            for (int e = 0; e < 8; ++e) sVt[ldcol + e][row] = vv[e];
        }
        wait_async0();
        __syncthreads();

        // S = Q K^T  (16 q-rows x 64 kv-cols per wave)
        v8f s[4];
#pragma unroll
        for (int nt = 0; nt < 4; ++nt) s[nt] = vzero;
#pragma unroll
        for (int ks = 0; ks < 2; ++ks) {
            v16bf a;
            int row = wv * 16 + l15;
            int klo = hlf * 8;
#pragma unroll
            for (int v = 0; v < 8; ++v) {
                int kk = ks * 32 + klo * 2 + 2 * v;
                a[2 * v]     = sQ[row][kk];
                a[2 * v + 1] = sQ[row][kk + 1];
            }
#pragma unroll
            for (int nt = 0; nt < 4; ++nt) {
                v16bf bb;
                int cn = nt * 16 + l15;
                int kb0 = hlf * 16 + ks * 32;
#pragma unroll
                for (int v = 0; v < 8; ++v) {
                    bb[2 * v]     = sK[cn][kb0 + 2 * v];
                    bb[2 * v + 1] = sK[cn][kb0 + 2 * v + 1];
                }
                s[nt] = __builtin_amdgcn_wmma_f32_16x16x32_bf16(
                    false, a, false, bb, (short)0, s[nt], false, false);
            }
        }

        // online softmax per q-row; write P tile to LDS (same-wave rows only)
#pragma unroll
        for (int r = 0; r < 8; ++r) {
            int qg = q0 + wv * 16 + r + hlf * 8;
            float rm = -1e30f;
#pragma unroll
            for (int nt = 0; nt < 4; ++nt) {
                float val = s[nt][r] * scale;
                int kg = k0 + nt * 16 + l15;
                if (kg > qg) val = -1e30f;   // causal mask
                s[nt][r] = val;
                rm = fmaxf(rm, val);
            }
            for (int off = 8; off >= 1; off >>= 1) rm = fmaxf(rm, __shfl_xor(rm, off, 16));
            float newm = fmaxf(mr[r], rm);
            float alpha = __expf(mr[r] - newm);
            mr[r] = newm;
            float rs = 0.f;
            int prow = wv * 16 + r + hlf * 8;
#pragma unroll
            for (int nt = 0; nt < 4; ++nt) {
                float p = __expf(s[nt][r] - newm);
                rs += p;
                sP[prow][nt * 16 + l15] = (__bf16)p;
            }
            for (int off = 8; off >= 1; off >>= 1) rs += __shfl_xor(rs, off, 16);
            lr[r] = lr[r] * alpha + rs;
#pragma unroll
            for (int nt = 0; nt < 4; ++nt) oacc[nt][r] *= alpha;
        }

        // O += P V   (A = P 16x64; B-frag from sVt: contiguous along kv)
#pragma unroll
        for (int ks = 0; ks < 2; ++ks) {
            v16bf a;
            int row = wv * 16 + l15;
            int klo = hlf * 8;
#pragma unroll
            for (int v = 0; v < 8; ++v) {
                int kk = ks * 32 + klo * 2 + 2 * v;
                a[2 * v]     = sP[row][kk];
                a[2 * v + 1] = sP[row][kk + 1];
            }
#pragma unroll
            for (int nt = 0; nt < 4; ++nt) {
                v16bf bb;
                int cn = nt * 16 + l15;            // head-dim column
                int kb0 = hlf * 16 + ks * 32;      // kv index
#pragma unroll
                for (int v = 0; v < 8; ++v) {
                    bb[2 * v]     = sVt[cn][kb0 + 2 * v];
                    bb[2 * v + 1] = sVt[cn][kb0 + 2 * v + 1];
                }
                oacc[nt] = __builtin_amdgcn_wmma_f32_16x16x32_bf16(
                    false, a, false, bb, (short)0, oacc[nt], false, false);
            }
        }
        __syncthreads();
    }

    // write O / l  -> [B,T,D] bf16 (head-contiguous)
#pragma unroll
    for (int nt = 0; nt < 4; ++nt)
#pragma unroll
        for (int r = 0; r < 8; ++r) {
            int qrow = q0 + wv * 16 + r + hlf * 8;
            float val = oacc[nt][r] / lr[r];
            o[((size_t)(b * TT + qrow)) * DD + h * DKK + nt * 16 + l15] = (__bf16)val;
        }
}

// ---------------- orchestration ----------------
extern "C" void kernel_launch(void* const* d_in, const int* in_sizes, int n_in,
                              void* d_out, int out_size, void* d_ws, size_t ws_size,
                              hipStream_t stream) {
    (void)in_sizes; (void)n_in; (void)out_size; (void)ws_size;
    const int*   ids         = (const int*)d_in[0];
    const float* wte         = (const float*)d_in[1];
    const float* wpe         = (const float*)d_in[2];
    const float* c_attn_w    = (const float*)d_in[3];
    const float* attn_proj_w = (const float*)d_in[4];
    const float* ln1_g       = (const float*)d_in[5];
    const float* ln1_b       = (const float*)d_in[6];
    const float* ln2_g       = (const float*)d_in[7];
    const float* ln2_b       = (const float*)d_in[8];
    const float* fc_w        = (const float*)d_in[9];
    const float* fc_b        = (const float*)d_in[10];
    const float* proj_w      = (const float*)d_in[11];
    const float* proj_b      = (const float*)d_in[12];
    const float* lnf_g       = (const float*)d_in[13];
    const float* lnf_b       = (const float*)d_in[14];
    float* logits = (float*)d_out;

    char* wsp = (char*)d_ws;
    auto alloc = [&](size_t bytes) -> char* {
        char* p = wsp;
        wsp += (bytes + 255) & ~(size_t)255;
        return p;
    };
    __bf16* wte_bf   = (__bf16*)alloc((size_t)VV * DD * 2);
    __bf16* cattn_bf = (__bf16*)alloc((size_t)LL * 3 * DD * DD * 2);
    __bf16* aproj_bf = (__bf16*)alloc((size_t)LL * DD * DD * 2);
    __bf16* fc_bf    = (__bf16*)alloc((size_t)LL * DFF * DD * 2);
    __bf16* proj_bf  = (__bf16*)alloc((size_t)LL * DD * DFF * 2);
    float*  x        = (float*)alloc((size_t)MTOK * DD * 4);
    __bf16* xn       = (__bf16*)alloc((size_t)MTOK * DD * 2);
    __bf16* qkv      = (__bf16*)alloc((size_t)MTOK * 3 * DD * 2);
    __bf16* obuf     = (__bf16*)alloc((size_t)MTOK * DD * 2);
    __bf16* hbuf     = (__bf16*)alloc((size_t)MTOK * DFF * 2);

    // weights -> bf16
    cvt_f32_bf16_kernel<<<8192, 256, 0, stream>>>(wte, wte_bf, (long)VV * DD);
    cvt_f32_bf16_kernel<<<4096, 256, 0, stream>>>(c_attn_w, cattn_bf, (long)LL * 3 * DD * DD);
    cvt_f32_bf16_kernel<<<2048, 256, 0, stream>>>(attn_proj_w, aproj_bf, (long)LL * DD * DD);
    cvt_f32_bf16_kernel<<<4096, 256, 0, stream>>>(fc_w, fc_bf, (long)LL * DFF * DD);
    cvt_f32_bf16_kernel<<<4096, 256, 0, stream>>>(proj_w, proj_bf, (long)LL * DD * DFF);

    // x = wte[ids] + wpe
    embed_kernel<<<(MTOK * DD) / 256, 256, 0, stream>>>(ids, wte, wpe, x);

    for (int l = 0; l < LL; ++l) {
        layernorm_bf16_kernel<<<MTOK, 256, 0, stream>>>(x, ln1_g + l * DD, ln1_b + l * DD, xn);
        // qkv = xn @ c_attn^T   [4096 x 3072]
        gemm_bf16_wmma<0><<<dim3((3 * DD) / 128, MTOK / 128), 256, 0, stream>>>(
            xn, cattn_bf + (size_t)l * 3 * DD * DD, nullptr, nullptr, qkv, MTOK, 3 * DD, DD);
        // causal MHA
        flash_attn_kernel<<<dim3(TT / 64, BB * HH), 128, 0, stream>>>(qkv, obuf);
        // x += o @ attn_proj^T
        gemm_bf16_wmma<1><<<dim3(DD / 128, MTOK / 128), 256, 0, stream>>>(
            obuf, aproj_bf + (size_t)l * DD * DD, nullptr, x, nullptr, MTOK, DD, DD);
        layernorm_bf16_kernel<<<MTOK, 256, 0, stream>>>(x, ln2_g + l * DD, ln2_b + l * DD, xn);
        // h = gelu(xn @ fc^T + fc_b)
        gemm_bf16_wmma<2><<<dim3(DFF / 128, MTOK / 128), 256, 0, stream>>>(
            xn, fc_bf + (size_t)l * DFF * DD, fc_b + (size_t)l * DFF, nullptr, hbuf, MTOK, DFF, DD);
        // x += h @ proj^T + proj_b
        gemm_bf16_wmma<3><<<dim3(DD / 128, MTOK / 128), 256, 0, stream>>>(
            hbuf, proj_bf + (size_t)l * DD * DFF, proj_b + (size_t)l * DD, x, nullptr, MTOK, DD, DFF);
    }

    layernorm_bf16_kernel<<<MTOK, 256, 0, stream>>>(x, lnf_g, lnf_b, xn);
    // logits = xn @ wte^T   [4096 x 32000]
    gemm_bf16_wmma<4><<<dim3(VV / 128, MTOK / 128), 256, 0, stream>>>(
        xn, wte_bf, nullptr, logits, nullptr, MTOK, VV, DD);
}
